// MultiHeadAttention_32238024524256
// MI455X (gfx1250) — matrix operand order repaired and checked by reading, not compiled, not run on hardware
//
#include <hip/hip_runtime.h>
#include <stdint.h>

// ---------------------------------------------------------------------------
// Fused MHA for MI455X (gfx1250, wave32, WMMA bf16 16x16x32, f32 accumulate)
//   k1: QKV projection GEMMs  (f32 in -> bf16 Q/K/V in workspace)
//   k2: flash-attention, TDM double-buffered K tiles, exact JAX-threefry
//       block dropout -> bf16 ctx
//   k3: output projection GEMM + bias -> f32 out
// ---------------------------------------------------------------------------

constexpr int B_  = 2;
constexpr int T_  = 2112;   // NX + NLQ
constexpr int NX_ = 2048;
constexpr int D_  = 768;
constexpr int H_  = 12;
constexpr int HD_ = 64;
constexpr int M_  = B_ * T_;   // 4224 rows total

typedef __attribute__((ext_vector_type(16))) __bf16 v16bf;
typedef __attribute__((ext_vector_type(8)))  float  v8f;
typedef __attribute__((ext_vector_type(4)))  unsigned int u32x4;
typedef __attribute__((ext_vector_type(8)))  int i32x8;
typedef __attribute__((ext_vector_type(4)))  int i32x4;

union Frag { unsigned u[8]; v16bf v; };

// native gfx1250 f32->bf16 convert (single v_cvt, RNE)
__device__ __forceinline__ unsigned short f2bf(float f) {
  return __builtin_bit_cast(unsigned short, (__bf16)f);
}
__device__ __forceinline__ unsigned pack2(float a, float b) {
  return (unsigned)f2bf(a) | ((unsigned)f2bf(b) << 16);
}

// A-fragment K index map (16-bit A 16x32, ISA 7.12.2)
__device__ __forceinline__ int kmapA(int i, int lh) {
  return ((i & 4) << 2) + (lh << 3) + ((i & 3) << 1);
}
// B-fragment K index map (same K order as A: lanes 0-15 K=0..7,16..23; 16-31 K=8..15,24..31)
__device__ __forceinline__ int kmapB(int i, int lh) {
  return ((i & 4) << 2) + (lh << 3) + ((i & 3) << 1);
}

__device__ __forceinline__ v8f wmma_bf16(v16bf a, v16bf b, v8f c) {
  return __builtin_amdgcn_wmma_f32_16x16x32_bf16(false, a, false, b,
                                                 (short)0, c, false, false);
}

// JAX threefry2x32 with key(42) = (0, 42); 20 rounds, 5 key injections
__device__ __forceinline__ unsigned threefry_lo(unsigned c0, unsigned c1) {
  const unsigned k0 = 0u, k1 = 42u, k2 = 0x1BD11BDAu ^ 0u ^ 42u;
  unsigned x0 = c0 + k0, x1 = c1 + k1;
#define TF_R(r) { x0 += x1; x1 = (x1 << (r)) | (x1 >> (32 - (r))); x1 ^= x0; }
  TF_R(13) TF_R(15) TF_R(26) TF_R(6)  x0 += k1; x1 += k2 + 1u;
  TF_R(17) TF_R(29) TF_R(16) TF_R(24) x0 += k2; x1 += k0 + 2u;
  TF_R(13) TF_R(15) TF_R(26) TF_R(6)  x0 += k0; x1 += k1 + 3u;
  TF_R(17) TF_R(29) TF_R(16) TF_R(24) x0 += k1; x1 += k2 + 4u;
  TF_R(13) TF_R(15) TF_R(26) TF_R(6)  x0 += k2; x1 += k0 + 5u;
#undef TF_R
  return x1;   // low word of 64-bit threefry output (32-bit bits path)
}
__device__ __forceinline__ float u01(unsigned bits) {
  return __uint_as_float((bits >> 9) | 0x3F800000u) - 1.0f;
}

// ---------------------------------------------------------------------------
// TDM: 2-D tile load, 32 rows x 64 bf16, row stride 768 elements, with LDS
// padding of 2 dwords after every 32 dwords (one row) -> 68-half LDS stride.
// D# bitfields per CDNA5 ISA ch.8 (groups 0/1; groups 2/3 zero for 2-D).
// amdgpu-toolchain (clang-23) 6-arg builtin form.
// ---------------------------------------------------------------------------
__device__ __forceinline__ void tdm_load_tile32x64(const unsigned short* gsrc,
                                                   unsigned lds_off) {
  unsigned long long ga = (unsigned long long)(uintptr_t)gsrc;
  u32x4 g0;
  g0[0] = 1u;                                        // count=1 (valid), user
  g0[1] = lds_off;                                   // lds_addr (bytes)
  g0[2] = (unsigned)(ga & 0xFFFFFFFFu);              // global_addr[31:0]
  g0[3] = (unsigned)((ga >> 32) & 0x01FFFFFFu)       // global_addr[56:32]
          | (2u << 30);                              // type=2 ("image")
  i32x8 g1;
  g1[0] = (int)((1u << 16)      // data_size = 2 bytes
              | (1u << 20)      // pad_enable
              | (4u << 22)      // pad_interval: 32 dwords (one 64-half row)
              | (1u << 25));    // pad_amount: 2 dwords (4 halves)
  g1[1] = (int)(768u << 16);    // tensor_dim0[15:0]=768 (bits 63:48)
  g1[2] = (int)(0x8000u << 16); // tensor_dim1[15:0]=32768 (bits 95:80)
  g1[3] = (int)(64u << 16);     // tile_dim0 = 64 (bits 127:112)
  g1[4] = 32;                   // tile_dim1 = 32, tile_dim2 = 0
  g1[5] = 768;                  // tensor_dim0_stride[31:0] = 768 elements
  g1[6] = 0;
  g1[7] = 0;
  i32x4 z4 = {0, 0, 0, 0};
  i32x8 z8 = {0, 0, 0, 0, 0, 0, 0, 0};
  __builtin_amdgcn_tensor_load_to_lds(g0, g1, z4, z4, z8, 0);
}

// ---------------------------------------------------------------------------
// Kernel 1: [concat(x,lq)] @ W -> bf16 dst.  grid (33,12), block 256 (8 waves)
// WG tile 128x64, wave tile 32x32 (2x2 WMMA), K step 32.
// ---------------------------------------------------------------------------
__global__ __launch_bounds__(256)
void qkv_gemm(const float* __restrict__ x, const float* __restrict__ lq,
              const float* __restrict__ W, unsigned short* __restrict__ dst) {
  __shared__ unsigned short As[128][36];   // [m][k] bf16, padded stride
  __shared__ unsigned short Bs[64][36];    // [n][k] = W[k][n] bf16

  const int tid   = threadIdx.x;
  const int lane  = tid & 31, laneh = lane >> 4, lanem = lane & 15;
  const int w     = tid >> 5;
  const int mwave = w & 3, nwave = w >> 2;
  const int mBase = blockIdx.x * 128, nBase = blockIdx.y * 64;

  // per-thread A row (concat source)
  const int r  = tid >> 1;
  const int kc = (tid & 1) * 16;
  const int gm = mBase + r;
  const int bb = gm / T_, tt = gm - bb * T_;
  const float* rowp = (tt < NX_)
      ? (x  + ((size_t)bb * NX_ + tt) * D_)
      : (lq + ((size_t)bb * (T_ - NX_) + (tt - NX_)) * D_);

  v8f acc[2][2] = {};

  for (int k0 = 0; k0 < D_; k0 += 32) {
    // ---- A tile: 128x32 f32 -> bf16, coalesced float4 loads
#pragma unroll
    for (int q = 0; q < 4; ++q) {
      float4 f = *(const float4*)(rowp + k0 + kc + 4 * q);
      uint2 p; p.x = pack2(f.x, f.y); p.y = pack2(f.z, f.w);
      *(uint2*)&As[r][kc + 4 * q] = p;
    }
    // ---- B tile transposed: Bs[n][k] = W[k0+k][nBase+n]
#pragma unroll
    for (int it = 0; it < 8; ++it) {
      int idx = tid + it * 256;
      int n = idx >> 5, k = idx & 31;
      Bs[n][k] = f2bf(W[(size_t)(k0 + k) * D_ + nBase + n]);
    }
    if (k0 + 32 < D_) {   // prefetch next K-step tiles (global_prefetch_b8)
      __builtin_prefetch(rowp + k0 + 32 + kc, 0, 3);
      __builtin_prefetch(&W[(size_t)(k0 + 32 + (tid & 31)) * D_ + nBase +
                            (tid >> 5) * 8], 0, 3);
    }
    __syncthreads();

    Frag af[2], bf[2];
#pragma unroll
    for (int mi = 0; mi < 2; ++mi) {
      int row = mwave * 32 + mi * 16 + lanem;
#pragma unroll
      for (int i = 0; i < 8; ++i)
        af[mi].u[i] = *(const unsigned*)&As[row][kmapA(i, laneh)];
    }
#pragma unroll
    for (int ni = 0; ni < 2; ++ni) {
      int col = nwave * 32 + ni * 16 + lanem;
#pragma unroll
      for (int i = 0; i < 8; ++i)
        bf[ni].u[i] = *(const unsigned*)&Bs[col][kmapB(i, laneh)];
    }
#pragma unroll
    for (int mi = 0; mi < 2; ++mi)
#pragma unroll
      for (int ni = 0; ni < 2; ++ni)
        acc[mi][ni] = wmma_bf16(af[mi].v, bf[ni].v, acc[mi][ni]);
    __syncthreads();
  }

#pragma unroll
  for (int mi = 0; mi < 2; ++mi)
#pragma unroll
    for (int ni = 0; ni < 2; ++ni)
#pragma unroll
      for (int j = 0; j < 8; ++j) {
        int gm2 = mBase + mwave * 32 + mi * 16 + j + (laneh << 3);
        int gn  = nBase + nwave * 32 + ni * 16 + lanem;
        dst[(size_t)gm2 * D_ + gn] = f2bf(acc[mi][ni][j]);
      }
}

// ---------------------------------------------------------------------------
// Kernel 2: fused flash attention + threefry dropout.
// grid (33, 12, 2), block 128 (4 waves). Wave owns 16 query rows.
// K tiles stream via TDM, double-buffered against TENSORcnt.
// ---------------------------------------------------------------------------
__global__ __launch_bounds__(128)
void attn(const unsigned short* __restrict__ Qw,
          const unsigned short* __restrict__ Kw,
          const unsigned short* __restrict__ Vw,
          unsigned short* __restrict__ Cw) {
  __shared__ unsigned short Kt[2][32][68];  // double-buffered TDM dest
  __shared__ unsigned short Vt[64][36];     // [d][kk]  bf16 transposed
  __shared__ unsigned short Pb[4][16][36];  // per-wave P tile [row][kk]

  const int tid = threadIdx.x, lane = tid & 31, wave = tid >> 5;
  const int laneh = lane >> 4, lanem = lane & 15;
  const int qblk = blockIdx.x, h = blockIdx.y, b = blockIdx.z;
  const int t0 = qblk * 64 + wave * 16;
  constexpr int NSTEP = T_ / 32;            // 66

  // Q fragments (16 rows x 64 d) resident in registers for whole kernel
  Frag qf[2];
  {
    const unsigned short* qrow =
        Qw + ((size_t)(b * T_ + t0 + lanem)) * D_ + h * HD_;
#pragma unroll
    for (int f = 0; f < 2; ++f)
#pragma unroll
      for (int i = 0; i < 8; ++i)
        qf[f].u[i] = *(const unsigned*)&qrow[f * 32 + kmapA(i, laneh)];
  }

  float m[8], l[8];
#pragma unroll
  for (int j = 0; j < 8; ++j) { m[j] = -1e30f; l[j] = 0.f; }
  v8f acc[4] = {};

  const int krow = tid >> 2, kch = tid & 3;   // cooperative V loaders
  const unsigned short* kgbase = Kw + ((size_t)b * T_) * D_ + h * HD_;
  const unsigned kt_base   = (unsigned)(uintptr_t)&Kt[0][0][0];
  const unsigned kt_stride = 32u * 68u * 2u;   // bytes per buffer

  if (wave == 0)                              // K tile 0 -> buffer 0
    tdm_load_tile32x64(kgbase, kt_base);

  for (int it = 0; it < NSTEP; ++it) {
    const int kk0 = it * 32;
    __syncthreads();   // previous compute done: buffers reusable
    if (wave == 0 && it + 1 < NSTEP)          // stream next K tile
      tdm_load_tile32x64(kgbase + (size_t)(kk0 + 32) * D_,
                         kt_base + (unsigned)((it + 1) & 1) * kt_stride);
    {   // V rows -> Vt[d][kk] (transposed), overlaps with TDM in flight
      const uint4* gv = (const uint4*)(Vw +
          ((size_t)(b * T_ + kk0 + krow)) * D_ + h * HD_ + kch * 16);
      uint4 v0 = gv[0], v1 = gv[1];
      unsigned vv[8] = {v0.x, v0.y, v0.z, v0.w, v1.x, v1.y, v1.z, v1.w};
#pragma unroll
      for (int i = 0; i < 8; ++i) {
        Vt[kch * 16 + 2 * i][krow]     = (unsigned short)(vv[i] & 0xFFFFu);
        Vt[kch * 16 + 2 * i + 1][krow] = (unsigned short)(vv[i] >> 16);
      }
      if (it + 1 < NSTEP)
        __builtin_prefetch(Vw + ((size_t)(b * T_ + kk0 + 32 + krow)) * D_ +
                           h * HD_ + kch * 16, 0, 1);
    }
    if (it + 1 < NSTEP) __builtin_amdgcn_s_wait_tensorcnt((short)1);
    else                __builtin_amdgcn_s_wait_tensorcnt((short)0);
    __syncthreads();   // K tile (TDM) + V tile visible to all waves

    const unsigned short (*Kcur)[68] = Kt[it & 1];

    // ---- S = Q K^T for 32 kk columns (two 16x16 tiles)
    v8f s0 = {}, s1 = {};
    {
      Frag kf;
#pragma unroll
      for (int i = 0; i < 8; ++i)
        kf.u[i] = *(const unsigned*)&Kcur[lanem][kmapB(i, laneh)];
      s0 = wmma_bf16(qf[0].v, kf.v, s0);
#pragma unroll
      for (int i = 0; i < 8; ++i)
        kf.u[i] = *(const unsigned*)&Kcur[lanem][32 + kmapB(i, laneh)];
      s0 = wmma_bf16(qf[1].v, kf.v, s0);
#pragma unroll
      for (int i = 0; i < 8; ++i)
        kf.u[i] = *(const unsigned*)&Kcur[16 + lanem][kmapB(i, laneh)];
      s1 = wmma_bf16(qf[0].v, kf.v, s1);
#pragma unroll
      for (int i = 0; i < 8; ++i)
        kf.u[i] = *(const unsigned*)&Kcur[16 + lanem][32 + kmapB(i, laneh)];
      s1 = wmma_bf16(qf[1].v, kf.v, s1);
    }

    // ---- online softmax (denominator excludes dropout mask — exact)
    float corr[8];
#pragma unroll
    for (int j = 0; j < 8; ++j) {
      float a = s0[j] * 0.125f, c = s1[j] * 0.125f;   // 1/sqrt(64)
      float rm = fmaxf(a, c);
      rm = fmaxf(rm, __shfl_xor(rm, 1));
      rm = fmaxf(rm, __shfl_xor(rm, 2));
      rm = fmaxf(rm, __shfl_xor(rm, 4));
      rm = fmaxf(rm, __shfl_xor(rm, 8));
      float mn = fmaxf(m[j], rm);
      corr[j] = __expf(m[j] - mn);
      m[j] = mn;
      float p0 = __expf(a - mn), p1 = __expf(c - mn);
      s0[j] = p0; s1[j] = p1;
      float rs = p0 + p1;
      rs += __shfl_xor(rs, 1); rs += __shfl_xor(rs, 2);
      rs += __shfl_xor(rs, 4); rs += __shfl_xor(rs, 8);
      l[j] = l[j] * corr[j] + rs;
    }
#pragma unroll
    for (int n = 0; n < 4; ++n)
#pragma unroll
      for (int j = 0; j < 8; ++j) acc[n][j] *= corr[j];

    // ---- block dropout via threefry; write masked P (bf16) to LDS
#pragma unroll
    for (int tile = 0; tile < 2; ++tile)
#pragma unroll
      for (int j = 0; j < 8; ++j) {
        int tq = t0 + j + (laneh << 3);
        int tk = kk0 + tile * 16 + lanem;
        unsigned idx =
            ((unsigned)((b * H_ + h) * T_ + tq)) * (unsigned)T_ + (unsigned)tk;
        float u = u01(threefry_lo(0u, idx));
        float keep = (tq < NX_ && tk < NX_) ? 0.9f : 0.8f;
        float pv = tile ? s1[j] : s0[j];
        pv = (u < keep) ? pv : 0.0f;
        Pb[wave][j + (laneh << 3)][tile * 16 + lanem] = f2bf(pv);
      }
    __syncthreads();

    // ---- acc += P (16x32) x V (32x64)
    Frag pf;
#pragma unroll
    for (int i = 0; i < 8; ++i)
      pf.u[i] = *(const unsigned*)&Pb[wave][lanem][kmapA(i, laneh)];
#pragma unroll
    for (int n = 0; n < 4; ++n) {
      Frag vf;
#pragma unroll
      for (int i = 0; i < 8; ++i)
        vf.u[i] = *(const unsigned*)&Vt[n * 16 + lanem][kmapB(i, laneh)];
      acc[n] = wmma_bf16(pf.v, vf.v, acc[n]);
    }
  }

  const float dscale = (float)((double)T_ * T_ /
      ((double)NX_ * NX_ * 0.9 + ((double)T_ * T_ - (double)NX_ * NX_) * 0.8));
#pragma unroll
  for (int n = 0; n < 4; ++n)
#pragma unroll
    for (int j = 0; j < 8; ++j) {
      int t   = t0 + j + (laneh << 3);
      int col = h * HD_ + n * 16 + lanem;
      Cw[((size_t)(b * T_ + t)) * D_ + col] = f2bf(acc[n][j] * (dscale / l[j]));
    }
}

// ---------------------------------------------------------------------------
// Kernel 3: ctx(bf16) @ Wo + bo -> f32 out.  Same tiling as kernel 1.
// ---------------------------------------------------------------------------
__global__ __launch_bounds__(256)
void out_proj(const unsigned short* __restrict__ Ain,
              const float* __restrict__ W, const float* __restrict__ bias,
              float* __restrict__ out) {
  __shared__ unsigned short As[128][36];
  __shared__ unsigned short Bs[64][36];

  const int tid   = threadIdx.x;
  const int lane  = tid & 31, laneh = lane >> 4, lanem = lane & 15;
  const int w     = tid >> 5;
  const int mwave = w & 3, nwave = w >> 2;
  const int mBase = blockIdx.x * 128, nBase = blockIdx.y * 64;

  const int r  = tid >> 1;
  const int kc = (tid & 1) * 16;
  const unsigned short* rowp = Ain + (size_t)(mBase + r) * D_;

  v8f acc[2][2] = {};

  for (int k0 = 0; k0 < D_; k0 += 32) {
#pragma unroll
    for (int q = 0; q < 4; ++q)
      *(uint2*)&As[r][kc + 4 * q] = *(const uint2*)(rowp + k0 + kc + 4 * q);
#pragma unroll
    for (int it = 0; it < 8; ++it) {
      int idx = tid + it * 256;
      int n = idx >> 5, k = idx & 31;
      Bs[n][k] = f2bf(W[(size_t)(k0 + k) * D_ + nBase + n]);
    }
    if (k0 + 32 < D_) {
      __builtin_prefetch(rowp + k0 + 32 + kc, 0, 3);
      __builtin_prefetch(&W[(size_t)(k0 + 32 + (tid & 31)) * D_ + nBase +
                            (tid >> 5) * 8], 0, 3);
    }
    __syncthreads();

    Frag af[2], bf[2];
#pragma unroll
    for (int mi = 0; mi < 2; ++mi) {
      int row = mwave * 32 + mi * 16 + lanem;
#pragma unroll
      for (int i = 0; i < 8; ++i)
        af[mi].u[i] = *(const unsigned*)&As[row][kmapA(i, laneh)];
    }
#pragma unroll
    for (int ni = 0; ni < 2; ++ni) {
      int col = nwave * 32 + ni * 16 + lanem;
#pragma unroll
      for (int i = 0; i < 8; ++i)
        bf[ni].u[i] = *(const unsigned*)&Bs[col][kmapB(i, laneh)];
    }
#pragma unroll
    for (int mi = 0; mi < 2; ++mi)
#pragma unroll
      for (int ni = 0; ni < 2; ++ni)
        acc[mi][ni] = wmma_bf16(af[mi].v, bf[ni].v, acc[mi][ni]);
    __syncthreads();
  }

#pragma unroll
  for (int mi = 0; mi < 2; ++mi)
#pragma unroll
    for (int ni = 0; ni < 2; ++ni)
#pragma unroll
      for (int j = 0; j < 8; ++j) {
        int gm2 = mBase + mwave * 32 + mi * 16 + j + (laneh << 3);
        int gn  = nBase + nwave * 32 + ni * 16 + lanem;
        out[(size_t)gm2 * D_ + gn] = acc[mi][ni][j] + bias[gn];
      }
}

// ---------------------------------------------------------------------------
extern "C" void kernel_launch(void* const* d_in, const int* in_sizes, int n_in,
                              void* d_out, int out_size, void* d_ws,
                              size_t ws_size, hipStream_t stream) {
  (void)in_sizes; (void)n_in; (void)out_size; (void)ws_size;
  const float* x  = (const float*)d_in[0];
  const float* lq = (const float*)d_in[1];
  const float* Wq = (const float*)d_in[2];
  const float* Wk = (const float*)d_in[3];
  const float* Wv = (const float*)d_in[4];
  const float* Wo = (const float*)d_in[5];
  const float* bo = (const float*)d_in[6];

  unsigned short* qb = (unsigned short*)d_ws;        // bf16 workspace
  const size_t S = (size_t)M_ * D_;
  unsigned short* kb = qb + S;
  unsigned short* vb = kb + S;
  unsigned short* cb = vb + S;                       // ~26 MB total

  dim3 gg(M_ / 128, D_ / 64), bb(256);
  qkv_gemm<<<gg, bb, 0, stream>>>(x, lq, Wq, qb);
  qkv_gemm<<<gg, bb, 0, stream>>>(x, lq, Wk, kb);
  qkv_gemm<<<gg, bb, 0, stream>>>(x, lq, Wv, vb);
  attn<<<dim3(T_ / 64, H_, B_), dim3(128), 0, stream>>>(qb, kb, vb, cb);
  out_proj<<<gg, bb, 0, stream>>>(cb, Wo, bo, (float*)d_out);
}
